// GAT_Layer_73297911874087
// MI455X (gfx1250) — matrix operand order repaired and checked by reading, not compiled, not run on hardware
//
#include <hip/hip_runtime.h>
#include <hip/hip_bf16.h>

// GAT layer for MI455X (gfx1250, wave32, WMMA + async-to-LDS).
// N=8000 nodes, F=256 features.
constexpr int NN = 8000;
constexpr int FF = 256;
constexpr int NW = NN / 32;          // 250 mask words per row
#define NEG_SLOPE 0.2f

typedef __attribute__((ext_vector_type(2)))  float  v2f;
typedef __attribute__((ext_vector_type(8)))  float  v8f;
typedef __attribute__((ext_vector_type(4)))  int    v4i;
typedef __attribute__((ext_vector_type(16))) __bf16 v16bf;

union ABu  { uint4 u[2]; v16bf v; };                 // 32B WMMA A/B fragment
union HBu  { unsigned short s[8]; uint4 q; };        // 8 bf16 -> one 16B store

#if defined(__has_builtin)
#  if __has_builtin(__builtin_amdgcn_global_load_async_to_lds_b128) && \
      __has_builtin(__builtin_amdgcn_s_wait_asynccnt)
#    define HAS_ASYNC 1
#  endif
#endif
#ifndef HAS_ASYNC
#  define HAS_ASYNC 0
#endif

__device__ __forceinline__ unsigned f2bf1(float x) { // f32 -> bf16 (RNE-ish)
  union { float f; unsigned u; } c; c.f = x;
  return (c.u + 0x7FFFu + ((c.u >> 16) & 1u)) >> 16;
}
__device__ __forceinline__ unsigned packbf2(float a, float b) {
  return f2bf1(a) | (f2bf1(b) << 16);
}

// 16B global -> LDS async copy (ASYNCcnt); falls back to synchronous copy.
// Builtin wants (addrspace(1) v4i*, addrspace(3) v4i*, imm offset, imm cpol).
__device__ __forceinline__ void cp16_g2l(void* l, const void* g) {
#if HAS_ASYNC
  __builtin_amdgcn_global_load_async_to_lds_b128(
      (__attribute__((address_space(1))) v4i*)(unsigned long long)g,
      (__attribute__((address_space(3))) v4i*)(unsigned)(unsigned long long)l,
      0, 0);
#else
  *(uint4*)l = *(const uint4*)g;
#endif
}
__device__ __forceinline__ void wait_async() {
#if HAS_ASYNC
  __builtin_amdgcn_s_wait_asynccnt(0);
#endif
}

// ---------------------------------------------------------------------------
// Kernel 1: compress adj (256 MB int32) -> bitmask (8 MB). One wave per word.
// bit l of word w of row i  <->  adj[i][w*32 + l]
// ---------------------------------------------------------------------------
__global__ void k_packmask(const int* __restrict__ adj, unsigned* __restrict__ mask) {
  const int wv   = (blockIdx.x * blockDim.x + threadIdx.x) >> 5;   // global wave id
  const int lane = threadIdx.x & 31;
  const int row  = wv / NW;
  const int w    = wv - row * NW;
  const int a    = adj[row * NN + w * 32 + lane];                  // 128B coalesced
  unsigned m = __builtin_amdgcn_ballot_w32(a != 0);
  if (lane == 0) mask[wv] = m;
}

// ---------------------------------------------------------------------------
// Kernel 2: H = X @ W^T + b  via v_wmma_f32_16x16x4_f32.
// Also writes HbT[f][i] = bf16(H[i][f])  (transposed, for big-GEMM B frags).
// ---------------------------------------------------------------------------
__global__ void __launch_bounds__(256) k_gemm_h(
    const float* __restrict__ X, const float* __restrict__ Ww,
    const float* __restrict__ Wb, float* __restrict__ H,
    unsigned short* __restrict__ HbT) {
  const int wave = threadIdx.x >> 5;
  const int lane = threadIdx.x & 31;
  const int i0 = blockIdx.x * 16;
  const int f0 = (blockIdx.y * 8 + wave) * 16;
  const int m = lane & 15, h = lane >> 4;            // A: lane=M, halves carry K pairs

  v8f c;
  #pragma unroll
  for (int v = 0; v < 8; ++v) c[v] = 0.0f;

  // A[m][k] = X[i0+m][k] ; B[k][n] = (W^T)[k][n] = Ww[n][k]  (n == lane&15 == m)
  const v2f* pa = (const v2f*)(X  + (i0 + m) * FF) + h;
  const v2f* pb = (const v2f*)(Ww + (f0 + m) * FF) + h;
  #pragma unroll 4
  for (int kk = 0; kk < FF / 4; ++kk) {              // K=4 per WMMA, 64 steps
    v2f a = pa[2 * kk];                              // K = 4kk+2h, 4kk+2h+1
    v2f b = pb[2 * kk];
    c = __builtin_amdgcn_wmma_f32_16x16x4_f32(false, a, false, b,
                                              (short)0, c, false, false);
  }

  const float bias = Wb[f0 + m];                     // C layout: lane=N -> same f
  HBu hb;
  #pragma unroll
  for (int v = 0; v < 8; ++v) {                      // rows M = 8h + v
    float val = c[v] + bias;
    H[(i0 + 8 * h + v) * FF + f0 + m] = val;
    hb.s[v] = (unsigned short)f2bf1(val);
  }
  *(uint4*)(HbT + (f0 + m) * NN + i0 + 8 * h) = hb.q;  // 8 consecutive i -> 16B store
}

// ---------------------------------------------------------------------------
// Kernel 3: s[i] = H[i].a_src + a_b ;  d[i] = H[i].a_dst.  One wave per row.
// ---------------------------------------------------------------------------
__global__ void k_rowdots(const float* __restrict__ H, const float* __restrict__ asrc,
                          const float* __restrict__ adst, const float* __restrict__ ab,
                          float* __restrict__ sv, float* __restrict__ dvv) {
  const int row  = blockIdx.x * 8 + (threadIdx.x >> 5);
  const int lane = threadIdx.x & 31;
  float as = 0.f, ad = 0.f;
  const float* hr = H + row * FF;
  #pragma unroll
  for (int k = lane; k < FF; k += 32) {
    float hv = hr[k];
    as += hv * asrc[k];
    ad += hv * adst[k];
  }
  #pragma unroll
  for (int off = 16; off > 0; off >>= 1) {
    as += __shfl_xor(as, off, 32);
    ad += __shfl_xor(ad, off, 32);
  }
  if (lane == 0) { sv[row] = as + ab[0]; dvv[row] = ad; }
}

// ---------------------------------------------------------------------------
// Kernel 4: per-row softmax stats over masked leakyrelu(s_i + d_j).
// One wave per row, lane l scans columns j = 32w + l. Online (m, l) + merge.
// ---------------------------------------------------------------------------
__global__ void k_rowstats(const float* __restrict__ sv, const float* __restrict__ dvv,
                           const unsigned* __restrict__ mask,
                           float* __restrict__ mrow, float* __restrict__ rrow) {
  const int row  = blockIdx.x * 8 + (threadIdx.x >> 5);
  const int lane = threadIdx.x & 31;
  const float sr = sv[row];
  float m = -__builtin_inff(), l = 0.f;
  for (int w = 0; w < NW; ++w) {
    float e = sr + dvv[w * 32 + lane];
    e = (e >= 0.f) ? e : NEG_SLOPE * e;
    if ((mask[row * NW + w] >> lane) & 1u) {
      if (e > m) { l = l * __expf(m - e) + 1.0f; m = e; }   // exp(-inf)=0 first hit
      else       { l += __expf(e - m); }
    }
  }
  #pragma unroll
  for (int off = 16; off > 0; off >>= 1) {
    float m2 = __shfl_xor(m, off, 32);
    float l2 = __shfl_xor(l, off, 32);
    float nm = fmaxf(m, m2);
    float la = (m  == nm) ? l  : l  * __expf(m  - nm);      // avoids inf-inf NaN
    float lb = (m2 == nm) ? l2 : l2 * __expf(m2 - nm);
    m = nm; l = la + lb;
  }
  if (lane == 0) { mrow[row] = m; rrow[row] = (l > 0.f) ? 1.0f / l : 0.0f; }
}

// ---------------------------------------------------------------------------
// Kernel 5: out = softmax(e) @ H, flash-style with precomputed (m_i, 1/l_i).
// Block = 256 thr (8 waves) owns 32 rows x 256 cols. Loop j in 32-wide tiles.
// Double-buffered pipeline, ONE barrier per tile:
//   - wave-private B tile (32 f-rows x 64B of HbT) staged one tile ahead via
//     global_load_async_to_lds_b128 (ASYNCcnt), s_wait_asynccnt before barrier
//   - P(32x32) bf16 tile for the NEXT iteration built by all 256 threads into
//     the other LDS buffer while WMMAs consume the current one
//   - 4x v_wmma_f32_16x16x32_bf16 per wave per tile (2 row-tiles x 2 f-tiles)
// ---------------------------------------------------------------------------
__global__ void __launch_bounds__(256) k_attn(
    const float* __restrict__ sv, const float* __restrict__ dvv,
    const float* __restrict__ mrow, const float* __restrict__ rrow,
    const unsigned* __restrict__ mask, const unsigned short* __restrict__ HbT,
    float* __restrict__ out) {
  const int t = threadIdx.x;
  const int wave = t >> 5, lane = t & 31;
  const int i0 = blockIdx.x * 32;

  __shared__ __align__(16) unsigned ldsP[2][32 * 20];     // 2 x 2560B P tiles
  __shared__ __align__(16) unsigned ldsB[2][8][32 * 20];  // 2 x 8 waves x 2560B B tiles
  __shared__ float ldsR[32], ldsM[32], ldsS[32];
  if (t < 32) { ldsR[t] = rrow[i0 + t]; ldsM[t] = mrow[i0 + t]; ldsS[t] = sv[i0 + t]; }
  __syncthreads();

  v8f acc[2][2];
  #pragma unroll
  for (int x = 0; x < 2; ++x)
    #pragma unroll
    for (int y = 0; y < 2; ++y)
      #pragma unroll
      for (int v = 0; v < 8; ++v) acc[x][y][v] = 0.f;

  const int r = lane;                    // phase-1: lane = row within block
  const int g = wave;                    // phase-1: wave = 4-col group
  const float srow = ldsS[r], mr = ldsM[r];
  const int lm = lane & 15, lh = lane >> 4;
  const float4* d4 = (const float4*)dvv;
  const unsigned* maskRow = mask + (i0 + r) * NW;
  // Async B staging: lane stages f-row (wave*32 + lane), 4 x 16B chunks.
  const unsigned short* gB = HbT + (wave * 32 + lane) * NN;

  // ---- prologue: stage tile 0 ----
  {
    unsigned* lB = &ldsB[0][wave][lane * 20];
    #pragma unroll
    for (int q = 0; q < 4; ++q) cp16_g2l(lB + q * 4, gB + q * 8);

    unsigned word = maskRow[0];
    float4 dq = d4[g];
    float e0 = srow + dq.x; e0 = (e0 >= 0.f) ? e0 : NEG_SLOPE * e0;
    float e1 = srow + dq.y; e1 = (e1 >= 0.f) ? e1 : NEG_SLOPE * e1;
    float e2 = srow + dq.z; e2 = (e2 >= 0.f) ? e2 : NEG_SLOPE * e2;
    float e3 = srow + dq.w; e3 = (e3 >= 0.f) ? e3 : NEG_SLOPE * e3;
    float p0 = ((word >> (g * 4 + 0)) & 1u) ? __expf(e0 - mr) : 0.f;
    float p1 = ((word >> (g * 4 + 1)) & 1u) ? __expf(e1 - mr) : 0.f;
    float p2 = ((word >> (g * 4 + 2)) & 1u) ? __expf(e2 - mr) : 0.f;
    float p3 = ((word >> (g * 4 + 3)) & 1u) ? __expf(e3 - mr) : 0.f;
    ldsP[0][r * 20 + g * 2 + 0] = packbf2(p0, p1);
    ldsP[0][r * 20 + g * 2 + 1] = packbf2(p2, p3);
  }
  wait_async();
  __syncthreads();

  for (int jt = 0; jt < NW; ++jt) {
    const int cur = jt & 1;
    // ---- prefetch next tile (uniform condition; overlaps with WMMAs) ----
    if (jt + 1 < NW) {
      const int jn = (jt + 1) * 32;
      unsigned* lB = &ldsB[cur ^ 1][wave][lane * 20];
      #pragma unroll
      for (int q = 0; q < 4; ++q) cp16_g2l(lB + q * 4, gB + jn + q * 8);

      unsigned word = maskRow[jt + 1];
      float4 dq = d4[(jt + 1) * 8 + g];
      float e0 = srow + dq.x; e0 = (e0 >= 0.f) ? e0 : NEG_SLOPE * e0;
      float e1 = srow + dq.y; e1 = (e1 >= 0.f) ? e1 : NEG_SLOPE * e1;
      float e2 = srow + dq.z; e2 = (e2 >= 0.f) ? e2 : NEG_SLOPE * e2;
      float e3 = srow + dq.w; e3 = (e3 >= 0.f) ? e3 : NEG_SLOPE * e3;
      float p0 = ((word >> (g * 4 + 0)) & 1u) ? __expf(e0 - mr) : 0.f;
      float p1 = ((word >> (g * 4 + 1)) & 1u) ? __expf(e1 - mr) : 0.f;
      float p2 = ((word >> (g * 4 + 2)) & 1u) ? __expf(e2 - mr) : 0.f;
      float p3 = ((word >> (g * 4 + 3)) & 1u) ? __expf(e3 - mr) : 0.f;
      ldsP[cur ^ 1][r * 20 + g * 2 + 0] = packbf2(p0, p1);
      ldsP[cur ^ 1][r * 20 + g * 2 + 1] = packbf2(p2, p3);
    }

    // ---- consume current tile: 4 WMMAs ----
    ABu afr[2];                          // A frag: lane lm row, lh selects K halves
    #pragma unroll
    for (int rt = 0; rt < 2; ++rt) {
      const unsigned* base = &ldsP[cur][(rt * 16 + lm) * 20 + lh * 4];
      afr[rt].u[0] = *(const uint4*)(base);        // K pairs {8lh..8lh+7}
      afr[rt].u[1] = *(const uint4*)(base + 8);    // K pairs {16+8lh..16+8lh+7}
    }
    const unsigned* bb = ldsB[cur][wave];
    #pragma unroll
    for (int ff = 0; ff < 2; ++ff) {               // B frag: lane=N row, lh = K half
      const unsigned* rowp = bb + (ff * 16 + lm) * 20 + lh * 4;
      ABu bfr; bfr.u[0] = *(const uint4*)rowp; bfr.u[1] = *(const uint4*)(rowp + 8);
      #pragma unroll
      for (int rt = 0; rt < 2; ++rt)
        acc[rt][ff] = __builtin_amdgcn_wmma_f32_16x16x32_bf16(
            false, afr[rt].v, false, bfr.v, (short)0, acc[rt][ff], false, false);
    }

    wait_async();          // own wave's next-B staged before anyone flips buffers
    __syncthreads();       // next-P visible; cur fully consumed before overwrite
  }

  // ---- epilogue: scale by 1/l_i and store ----
  #pragma unroll
  for (int rt = 0; rt < 2; ++rt)
    #pragma unroll
    for (int ff = 0; ff < 2; ++ff) {
      const int fcol = wave * 32 + ff * 16 + lm;
      #pragma unroll
      for (int v = 0; v < 8; ++v) {
        const int rr = rt * 16 + 8 * lh + v;       // C layout: M = 8*half + vgpr
        out[(i0 + rr) * FF + fcol] = acc[rt][ff][v] * ldsR[rr];
      }
    }
}

// ---------------------------------------------------------------------------
extern "C" void kernel_launch(void* const* d_in, const int* in_sizes, int n_in,
                              void* d_out, int out_size, void* d_ws, size_t ws_size,
                              hipStream_t stream) {
  (void)in_sizes; (void)n_in; (void)out_size; (void)ws_size;
  const float* X    = (const float*)d_in[0];   // [8000,256]
  const int*   adj  = (const int*)  d_in[1];   // [8000,8000]
  const float* Ww   = (const float*)d_in[2];   // [256,256]
  const float* Wb   = (const float*)d_in[3];   // [256]
  const float* asrc = (const float*)d_in[4];   // [256]
  const float* adst = (const float*)d_in[5];   // [256]
  const float* ab   = (const float*)d_in[6];   // [1]
  float* out = (float*)d_out;                  // [8000,256] f32

  char* ws = (char*)d_ws;                      // ~20.4 MB total, all 256B-aligned
  float*          H    = (float*)(ws);                    //  8,192,000 B
  unsigned short* HbT  = (unsigned short*)(ws + 8192000); //  4,096,000 B
  float*          sv   = (float*)(ws + 12288000);         //     32,000 B
  float*          dvv  = (float*)(ws + 12320000);         //     32,000 B
  float*          mrow = (float*)(ws + 12352000);         //     32,000 B
  float*          rrow = (float*)(ws + 12384000);         //     32,000 B
  unsigned*       maskp= (unsigned*)(ws + 12416000);      //  8,000,000 B

  k_packmask<<<(NN * NW) / 8, 256, 0, stream>>>(adj, maskp);
  k_gemm_h  <<<dim3(NN / 16, 2), 256, 0, stream>>>(X, Ww, Wb, H, HbT);
  k_rowdots <<<NN / 8, 256, 0, stream>>>(H, asrc, adst, ab, sv, dvv);
  k_rowstats<<<NN / 8, 256, 0, stream>>>(sv, dvv, maskp, mrow, rrow);
  k_attn    <<<NN / 32, 256, 0, stream>>>(sv, dvv, mrow, rrow, maskp, HbT, out);
}
